// LinearAttentionToeplitz_34617436406141
// MI455X (gfx1250) — matrix operand order repaired and checked by reading, not compiled, not run on hardware
//
#include <hip/hip_runtime.h>
#include <hip/hip_bf16.h>

// Problem constants
#define BB   8
#define TT   2048
#define EE   512
#define CH   64          // attention chunk length
#define NC   (TT / CH)   // 32 chunks per batch
#define PAN  64          // state column panel (serial fallback)
#define LDA  520         // padded LDS row (f16 elems), 1040B = 16B-aligned
#define LDC  72          // padded LDS row for 64-wide tiles, 144B = 16B-aligned

typedef __attribute__((ext_vector_type(16))) _Float16 v16h;
typedef __attribute__((ext_vector_type(8)))  _Float16 v8h;
typedef __attribute__((ext_vector_type(8)))  float    v8f;

__device__ __forceinline__ v8f vzero() {
    v8f a;
#pragma unroll
    for (int i = 0; i < 8; ++i) a[i] = 0.f;
    return a;
}

// A-fragment (16x32 f16, M x K) from row-major [m][k] storage, row stride ld.
// Lane m = L%16, chunks K = kb..kb+7 and kb+16..kb+23, kb = (L<16 ? 0 : 8).
__device__ __forceinline__ v16h frag_a(const _Float16* p0, int ld) {
    int lane = threadIdx.x & 31;
    const _Float16* p = p0 + (lane & 15) * ld + ((lane >> 4) << 3);
    v8h lo = *(const v8h*)(p);
    v8h hi = *(const v8h*)(p + 16);
    v16h r;
#pragma unroll
    for (int i = 0; i < 8; ++i) { r[i] = lo[i]; r[i + 8] = hi[i]; }
    return r;
}

// B-fragment (32x16 f16, K x N) from TRANSPOSED [n][k] row-major storage.
// Lane n = L%16, chunks K = kb..kb+7 and kb+16..kb+23, kb = (L<16 ? 0 : 8).
__device__ __forceinline__ v16h frag_b(const _Float16* p0, int ld) {
    int lane = threadIdx.x & 31;
    const _Float16* p = p0 + (lane & 15) * ld + ((lane >> 4) << 3);
    v8h lo = *(const v8h*)(p);
    v8h hi = *(const v8h*)(p + 16);
    v16h r;
#pragma unroll
    for (int i = 0; i < 8; ++i) { r[i] = lo[i]; r[i + 8] = hi[i]; }
    return r;
}

__device__ __forceinline__ v8f wmma_f16(v16h a, v16h b, v8f c) {
    return __builtin_amdgcn_wmma_f32_16x16x32_f16(false, a, false, b,
                                                  (short)0, c, false, false);
}

// ---------------------------------------------------------------------------
// Kernel 0: f32 -> f16 weight conversion
// ---------------------------------------------------------------------------
__global__ void cvt_f16_kernel(const float* __restrict__ src,
                               _Float16* __restrict__ dst, int n) {
    int i = blockIdx.x * 256 + threadIdx.x;
    if (i < n) dst[i] = (_Float16)src[i];
}

// ---------------------------------------------------------------------------
// Kernel 1: fused QKV projection + feature maps.
// q -> softmax * E^-1/2 ; k -> exp (written [t][e] and optionally [e][t]);
// v -> toeplitz scale.
// ---------------------------------------------------------------------------
__global__ __launch_bounds__(256)
void qkv_kernel(const float* __restrict__ x, const float* __restrict__ toep,
                const _Float16* __restrict__ wq, const _Float16* __restrict__ wk,
                const _Float16* __restrict__ wv,
                const float* __restrict__ qb, const float* __restrict__ kb,
                const float* __restrict__ vb,
                _Float16* __restrict__ qf, _Float16* __restrict__ kf,
                _Float16* __restrict__ vf, _Float16* __restrict__ kT) {
    __shared__ _Float16 xs[64 * LDA];   // x^T tile, [t][e]
    __shared__ _Float16 qs[64 * LDA];   // raw q staging for softmax

    int b  = blockIdx.x;
    int t0 = blockIdx.y * 64;
    int tid = threadIdx.x;

    for (int i = tid; i < 64 * EE; i += 256) {
        int ei = i >> 6, tl = i & 63;
        xs[tl * LDA + ei] = (_Float16)x[((size_t)b * EE + ei) * TT + t0 + tl];
    }
    __syncthreads();

    int w = tid >> 5, lane = tid & 31;
    int mt = w & 3;
    int ntBase = (w >> 2) * 48;     // 96 N-tiles: q[0,32) k[32,64) v[64,96)

    for (int nt = ntBase; nt < ntBase + 48; ++nt) {
        int which = nt >> 5;        // 0=q 1=k 2=v
        int col = (nt & 31) << 4;
        const _Float16* W = (which == 0) ? wq : (which == 1) ? wk : wv;
        const float* bias = (which == 0) ? qb : (which == 1) ? kb : vb;

        v8f acc;
        float bv = bias[col + (lane & 15)];
#pragma unroll
        for (int r = 0; r < 8; ++r) acc[r] = bv;

#pragma unroll
        for (int k = 0; k < EE; k += 32) {
            v16h a  = frag_a(xs + mt * 16 * LDA + k, LDA);
            v16h bm = frag_b(W + (size_t)col * EE + k, EE);
            acc = wmma_f16(a, bm, acc);
        }

        int nl = lane & 15, half = lane >> 4;
        int n = col + nl;
        if (which == 0) {
#pragma unroll
            for (int r = 0; r < 8; ++r)
                qs[(mt * 16 + r + half * 8) * LDA + n] = (_Float16)acc[r];
        } else if (which == 1) {
            v8h pk;
#pragma unroll
            for (int r = 0; r < 8; ++r) {
                float ev = __expf(acc[r]);
                pk[r] = (_Float16)ev;
                kf[((size_t)b * TT + t0 + mt * 16 + r + half * 8) * EE + n] =
                    (_Float16)ev;
            }
            if (kT)   // transposed copy [b][e][t], contiguous 16B store
                *(v8h*)&kT[((size_t)b * EE + n) * TT + t0 + mt * 16 + half * 8] = pk;
        } else {
#pragma unroll
            for (int r = 0; r < 8; ++r) {
                int t = t0 + mt * 16 + r + half * 8;
                vf[((size_t)b * TT + t) * EE + n] = (_Float16)(acc[r] * toep[t]);
            }
        }
    }
    __syncthreads();

    if (tid < 64) {
        const _Float16* row = qs + tid * LDA;
        float mx = -1e30f;
        for (int e = 0; e < EE; ++e) mx = fmaxf(mx, (float)row[e]);
        float s = 0.f;
        for (int e = 0; e < EE; ++e) s += __expf((float)row[e] - mx);
        float inv = 0.04419417382415922f / s;   // E^-0.5 / sum
        size_t base = ((size_t)b * TT + t0 + tid) * EE;
        for (int e = 0; e < EE; ++e)
            qf[base + e] = (_Float16)(__expf((float)row[e] - mx) * inv);
    }
}

// ---------------------------------------------------------------------------
// PARALLEL PATH, Pass A: per (batch, chunk) G_c = K_c^T V_c (stored as
// G^T[ev][e] so it is a ready-made B-operand) and zc_c = colsum(K_c).
// ---------------------------------------------------------------------------
__global__ __launch_bounds__(256)
void chunkg_kernel(const _Float16* __restrict__ kT,   // [b][e][t]
                   const _Float16* __restrict__ vf,   // [b][t][e]
                   _Float16* __restrict__ Gt,         // [b][chunk][ev][e]
                   float* __restrict__ zc) {          // [b][chunk][e]
    __shared__ _Float16 Kt[EE * LDC];   // [e][c]
    __shared__ _Float16 Vt[EE * LDC];   // [ev][c]
    int b = blockIdx.x, ch = blockIdx.y, c0 = ch * CH;
    int tid = threadIdx.x, w = tid >> 5, lane = tid & 31;
    int nl = lane & 15, half = lane >> 4;

    for (int i = tid; i < EE * (CH / 8); i += 256) {       // contiguous copies
        int e = i >> 3, g = i & 7;
        *(v8h*)&Kt[e * LDC + g * 8] =
            *(const v8h*)&kT[((size_t)b * EE + e) * TT + c0 + g * 8];
    }
    for (int i = tid; i < CH * EE; i += 256) {             // transpose V
        int tc = i >> 9, e = i & 511;
        Vt[e * LDC + tc] = vf[((size_t)b * TT + c0 + tc) * EE + e];
    }
    __syncthreads();

    _Float16* Gb = Gt + ((size_t)b * NC + ch) * EE * EE;
    for (int tIdx = w * 128; tIdx < w * 128 + 128; ++tIdx) {
        int mt = tIdx & 31, nt = tIdx >> 5;   // m = e-tile, n = ev-tile
        v8f acc = vzero();
#pragma unroll
        for (int k = 0; k < CH; k += 32) {
            v16h a  = frag_a(Kt + (mt * 16) * LDC + k, LDC);
            v16h bm = frag_b(Vt + (nt * 16) * LDC + k, LDC);
            acc = wmma_f16(a, bm, acc);
        }
        v8h pk;
#pragma unroll
        for (int r = 0; r < 8; ++r) pk[r] = (_Float16)acc[r];
        *(v8h*)&Gb[(size_t)(nt * 16 + nl) * EE + mt * 16 + half * 8] = pk;
    }
    for (int e = tid; e < EE; e += 256) {
        float s = 0.f;
        for (int c = 0; c < CH; ++c) s += (float)Kt[e * LDC + c];
        zc[((size_t)b * NC + ch) * EE + e] = s;
    }
}

// ---------------------------------------------------------------------------
// Pass B: in-place EXCLUSIVE prefix over chunks (vectorized 16B), plus zc.
// ---------------------------------------------------------------------------
__global__ __launch_bounds__(256)
void prefix_kernel(_Float16* __restrict__ Gt) {
    int b = blockIdx.x;
    size_t i = ((size_t)blockIdx.y * 256 + threadIdx.x) * 8;
    _Float16* base = Gt + (size_t)b * NC * EE * EE + i;
    v8f acc = vzero();
    for (int c = 0; c < NC; ++c) {
        v8h* p = (v8h*)(base + (size_t)c * EE * EE);
        v8h t = *p, o;
#pragma unroll
        for (int r = 0; r < 8; ++r) { o[r] = (_Float16)acc[r]; acc[r] += (float)t[r]; }
        *p = o;
    }
}

__global__ void zprefix_kernel(float* __restrict__ zc) {
    int b = blockIdx.x;
    int e = blockIdx.y * 256 + threadIdx.x;
    float* base = zc + (size_t)b * NC * EE + e;
    float acc = 0.f;
    for (int c = 0; c < NC; ++c) {
        float t = base[(size_t)c * EE];
        base[(size_t)c * EE] = acc;
        acc += t;
    }
}

// ---------------------------------------------------------------------------
// Pass C: per (batch, chunk): O = (Q @ S_prefix + mask(QK^T) @ V) / d,
// d = max(q . z_prefix + rowsum(masked A), eps). Fully parallel, no scan.
// ---------------------------------------------------------------------------
__global__ __launch_bounds__(256)
void chunkattn_kernel(const _Float16* __restrict__ qf, const _Float16* __restrict__ kf,
                      const _Float16* __restrict__ vf, const _Float16* __restrict__ Gt,
                      const float* __restrict__ zc, _Float16* __restrict__ attn) {
    __shared__ _Float16 Vt[EE * LDC];   // [ev][c]
    __shared__ _Float16 As[CH * LDC];   // masked QK^T
    __shared__ float dinv[CH];
    int b = blockIdx.x, ch = blockIdx.y, c0 = ch * CH;
    int tid = threadIdx.x, w = tid >> 5, lane = tid & 31;
    int nl = lane & 15, half = lane >> 4;
    const _Float16* Qb = qf + (size_t)b * TT * EE;
    const _Float16* Kb = kf + (size_t)b * TT * EE;
    const _Float16* Sp = Gt + ((size_t)b * NC + ch) * EE * EE; // S^T prefix [ev][e]
    const float*    zp = zc + ((size_t)b * NC + ch) * EE;

    for (int i = tid; i < CH * EE; i += 256) {
        int tc = i >> 9, e = i & 511;
        Vt[e * LDC + tc] = vf[((size_t)b * TT + c0 + tc) * EE + e];
    }
    // A = Q K^T with causal mask (writes As only; Vt untouched here)
    for (int tIdx = w * 2; tIdx < w * 2 + 2; ++tIdx) {
        int mi = tIdx >> 2, ni = tIdx & 3;
        v8f acc = vzero();
#pragma unroll
        for (int k = 0; k < EE; k += 32) {
            v16h a  = frag_a(Qb + (size_t)(c0 + mi * 16) * EE + k, EE);
            v16h bm = frag_b(Kb + (size_t)(c0 + ni * 16) * EE + k, EE);
            acc = wmma_f16(a, bm, acc);
        }
#pragma unroll
        for (int r = 0; r < 8; ++r) {
            int i_ = mi * 16 + r + half * 8, j_ = ni * 16 + nl;
            As[i_ * LDC + j_] = (j_ <= i_) ? (_Float16)acc[r] : (_Float16)0.f;
        }
    }
    __syncthreads();

    if (tid < CH) {
        float rs = 0.f;
        for (int j = 0; j < CH; ++j) rs += (float)As[tid * LDC + j];
        const _Float16* qrow = Qb + (size_t)(c0 + tid) * EE;
        float qz = 0.f;
        for (int e = 0; e < EE; ++e) qz += (float)qrow[e] * zp[e];
        dinv[tid] = 1.f / fmaxf(qz + rs, 1e-3f);
    }
    __syncthreads();

    // O over the full E columns: 4 x 32 tiles, 16 per wave
    for (int tIdx = w * 16; tIdx < w * 16 + 16; ++tIdx) {
        int mt = tIdx & 3, nt = tIdx >> 2;
        v8f acc = vzero();
#pragma unroll
        for (int k = 0; k < EE; k += 32) {
            v16h a  = frag_a(Qb + (size_t)(c0 + mt * 16) * EE + k, EE);
            v16h bm = frag_b(Sp + (size_t)(nt * 16) * EE + k, EE);
            acc = wmma_f16(a, bm, acc);
        }
#pragma unroll
        for (int k = 0; k < CH; k += 32) {
            v16h a  = frag_a(As + (mt * 16) * LDC + k, LDC);
            v16h bm = frag_b(Vt + (nt * 16) * LDC + k, LDC);
            acc = wmma_f16(a, bm, acc);
        }
#pragma unroll
        for (int r = 0; r < 8; ++r) {
            int i_ = mt * 16 + r + half * 8;
            attn[((size_t)b * TT + c0 + i_) * EE + nt * 16 + nl] =
                (_Float16)(acc[r] * dinv[i_]);
        }
    }
}

// ---------------------------------------------------------------------------
// SERIAL FALLBACK (small workspace): per (batch, 64-col panel) chunk scan.
// ---------------------------------------------------------------------------
__global__ __launch_bounds__(256)
void attn_serial_kernel(const _Float16* __restrict__ qf, const _Float16* __restrict__ kf,
                        const _Float16* __restrict__ vf, _Float16* __restrict__ attn) {
    __shared__ _Float16 St[PAN * LDA];
    __shared__ _Float16 Kt[EE * LDC];
    __shared__ _Float16 Vt[PAN * LDC];
    __shared__ _Float16 As[CH * LDC];
    __shared__ float z[EE];
    __shared__ float dinv[CH];

    int b = blockIdx.x, p0 = blockIdx.y * PAN;
    int tid = threadIdx.x, w = tid >> 5, lane = tid & 31;
    int nl = lane & 15, half = lane >> 4;

    for (int i = tid; i < PAN * LDA; i += 256) St[i] = (_Float16)0.f;
    for (int i = tid; i < EE; i += 256) z[i] = 0.f;
    __syncthreads();

    const _Float16* Qb = qf + (size_t)b * TT * EE;
    const _Float16* Kb = kf + (size_t)b * TT * EE;
    const _Float16* Vb = vf + (size_t)b * TT * EE;
    _Float16* Ob = attn + (size_t)b * TT * EE;

    for (int c0 = 0; c0 < TT; c0 += CH) {
        for (int i = tid; i < CH * EE; i += 256) {
            int tc = i >> 9, e = i & 511;
            Kt[e * LDC + tc] = Kb[(size_t)(c0 + tc) * EE + e];
        }
        for (int i = tid; i < CH * PAN; i += 256) {
            int tc = i >> 6, n = i & 63;
            Vt[n * LDC + tc] = Vb[(size_t)(c0 + tc) * EE + p0 + n];
        }
        __syncthreads();

        for (int tIdx = w * 2; tIdx < w * 2 + 2; ++tIdx) {
            int mi = tIdx >> 2, ni = tIdx & 3;
            v8f acc = vzero();
#pragma unroll
            for (int k = 0; k < EE; k += 32) {
                v16h a  = frag_a(Qb + (size_t)(c0 + mi * 16) * EE + k, EE);
                v16h bm = frag_b(Kb + (size_t)(c0 + ni * 16) * EE + k, EE);
                acc = wmma_f16(a, bm, acc);
            }
#pragma unroll
            for (int r = 0; r < 8; ++r) {
                int i_ = mi * 16 + r + half * 8, j_ = ni * 16 + nl;
                As[i_ * LDC + j_] = (j_ <= i_) ? (_Float16)acc[r] : (_Float16)0.f;
            }
        }
        __syncthreads();

        if (tid < CH) {
            float rs = 0.f;
            for (int j = 0; j < CH; ++j) rs += (float)As[tid * LDC + j];
            const _Float16* qrow = Qb + (size_t)(c0 + tid) * EE;
            float qz = 0.f;
            for (int e = 0; e < EE; ++e) qz += (float)qrow[e] * z[e];
            dinv[tid] = 1.f / fmaxf(qz + rs, 1e-3f);
        }
        __syncthreads();

        for (int tIdx = w * 2; tIdx < w * 2 + 2; ++tIdx) {
            int mi = tIdx >> 2, np = tIdx & 3;
            v8f acc = vzero();
#pragma unroll
            for (int k = 0; k < EE; k += 32) {
                v16h a  = frag_a(Qb + (size_t)(c0 + mi * 16) * EE + k, EE);
                v16h bm = frag_b(St + (np * 16) * LDA + k, LDA);
                acc = wmma_f16(a, bm, acc);
            }
#pragma unroll
            for (int k = 0; k < CH; k += 32) {
                v16h a  = frag_a(As + (mi * 16) * LDC + k, LDC);
                v16h bm = frag_b(Vt + (np * 16) * LDC + k, LDC);
                acc = wmma_f16(a, bm, acc);
            }
#pragma unroll
            for (int r = 0; r < 8; ++r) {
                int i_ = mi * 16 + r + half * 8;
                Ob[(size_t)(c0 + i_) * EE + p0 + np * 16 + nl] =
                    (_Float16)(acc[r] * dinv[i_]);
            }
        }
        __syncthreads();

        for (int tIdx = w * 16; tIdx < w * 16 + 16; ++tIdx) {
            int me = tIdx >> 2, nv = tIdx & 3;
            v8h cpk = *(const v8h*)&St[(nv * 16 + nl) * LDA + me * 16 + half * 8];
            v8f acc;
#pragma unroll
            for (int r = 0; r < 8; ++r) acc[r] = (float)cpk[r];
#pragma unroll
            for (int k = 0; k < CH; k += 32) {
                v16h a  = frag_a(Kt + (me * 16) * LDC + k, LDC);
                v16h bm = frag_b(Vt + (nv * 16) * LDC + k, LDC);
                acc = wmma_f16(a, bm, acc);
            }
            v8h pk;
#pragma unroll
            for (int r = 0; r < 8; ++r) pk[r] = (_Float16)acc[r];
            *(v8h*)&St[(nv * 16 + nl) * LDA + me * 16 + half * 8] = pk;
        }
        for (int e = tid; e < EE; e += 256) {
            float s = z[e];
            for (int c = 0; c < CH; ++c) s += (float)Kt[e * LDC + c];
            z[e] = s;
        }
        __syncthreads();
    }
}

// ---------------------------------------------------------------------------
// Kernel: output projection, f32 result.
// ---------------------------------------------------------------------------
__global__ __launch_bounds__(256)
void out_kernel(const _Float16* __restrict__ attn, const _Float16* __restrict__ wo,
                const float* __restrict__ ob, float* __restrict__ out) {
    int b = blockIdx.x, t0 = blockIdx.y * 64;
    int tid = threadIdx.x, w = tid >> 5, lane = tid & 31;
    int nl = lane & 15, half = lane >> 4;
    const _Float16* Ab = attn + ((size_t)b * TT + t0) * EE;

    for (int tIdx = w * 16; tIdx < w * 16 + 16; ++tIdx) {
        int mt = tIdx & 3, nt = tIdx >> 2;
        int col = nt * 16;
        v8f acc;
        float bv = ob[col + nl];
#pragma unroll
        for (int r = 0; r < 8; ++r) acc[r] = bv;
#pragma unroll
        for (int k = 0; k < EE; k += 32) {
            v16h a  = frag_a(Ab + mt * 16 * EE + k, EE);
            v16h bm = frag_b(wo + (size_t)col * EE + k, EE);
            acc = wmma_f16(a, bm, acc);
        }
#pragma unroll
        for (int r = 0; r < 8; ++r)
            out[((size_t)b * TT + t0 + mt * 16 + r + half * 8) * EE + col + nl] = acc[r];
    }
}

// ---------------------------------------------------------------------------
extern "C" void kernel_launch(void* const* d_in, const int* in_sizes, int n_in,
                              void* d_out, int out_size, void* d_ws, size_t ws_size,
                              hipStream_t stream) {
    (void)in_sizes; (void)n_in; (void)out_size;
    const float* x    = (const float*)d_in[0];
    const float* toep = (const float*)d_in[1];
    const float* qw   = (const float*)d_in[2];
    const float* qb   = (const float*)d_in[3];
    const float* kw   = (const float*)d_in[4];
    const float* kb   = (const float*)d_in[5];
    const float* vw   = (const float*)d_in[6];
    const float* vb   = (const float*)d_in[7];
    const float* ow   = (const float*)d_in[8];
    const float* ob   = (const float*)d_in[9];
    float* out = (float*)d_out;

    const size_t WTE = (size_t)EE * EE;          // weight elems
    const size_t SEQ = (size_t)BB * TT * EE;     // per-tensor elems
    const size_t GTE = (size_t)BB * NC * EE * EE;

    _Float16* wq = (_Float16*)d_ws;
    _Float16* wk = wq + WTE;
    _Float16* wv = wk + WTE;
    _Float16* wo = wv + WTE;
    _Float16* qf = wo + WTE;
    _Float16* kf = qf + SEQ;
    _Float16* vf = kf + SEQ;
    _Float16* at = vf + SEQ;
    _Float16* kT = at + SEQ;
    _Float16* Gt = kT + SEQ;
    float*    zc = (float*)(Gt + GTE);

    size_t parNeed = (4 * WTE + 5 * SEQ + GTE) * sizeof(_Float16) +
                     (size_t)BB * NC * EE * sizeof(float);
    bool par = ws_size >= parNeed;

    int wn = EE * EE, wb = (wn + 255) / 256;
    cvt_f16_kernel<<<wb, 256, 0, stream>>>(qw, wq, wn);
    cvt_f16_kernel<<<wb, 256, 0, stream>>>(kw, wk, wn);
    cvt_f16_kernel<<<wb, 256, 0, stream>>>(vw, wv, wn);
    cvt_f16_kernel<<<wb, 256, 0, stream>>>(ow, wo, wn);

    qkv_kernel<<<dim3(BB, TT / 64), 256, 0, stream>>>(
        x, toep, wq, wk, wv, qb, kb, vb, qf, kf, vf, par ? kT : (_Float16*)nullptr);

    if (par) {
        chunkg_kernel<<<dim3(BB, NC), 256, 0, stream>>>(kT, vf, Gt, zc);
        prefix_kernel<<<dim3(BB, (EE * EE) / 2048), 256, 0, stream>>>(Gt);
        zprefix_kernel<<<dim3(BB, EE / 256), 256, 0, stream>>>(zc);
        chunkattn_kernel<<<dim3(BB, NC), 256, 0, stream>>>(qf, kf, vf, Gt, zc, at);
    } else {
        attn_serial_kernel<<<dim3(BB, EE / PAN), 256, 0, stream>>>(qf, kf, vf, at);
    }

    out_kernel<<<dim3(BB, TT / 64), 256, 0, stream>>>(at, wo, ob, out);
}
